// RetentionUnit_30580167147624
// MI455X (gfx1250) — compile-verified, matrix operand order checked by reading, not run on hardware
//
#include <hip/hip_runtime.h>

typedef __attribute__((ext_vector_type(16))) _Float16 v16h;
typedef __attribute__((ext_vector_type(8)))  _Float16 v8h;
typedef __attribute__((ext_vector_type(8)))  float    v8f;

#define BB 2
#define TT 2048
#define EE 1024
#define HH 16
#define DD 64
#define LDA 72   // LDS leading dim (halfs): 144B rows, 16B-aligned, bank-staggered
#define LDF 72   // LDS leading dim (floats)
#define EPSV 1e-6f

// ---------------------------------------------------------------------------
// WMMA fragment load per CDNA5 ISA 16-bit A/B layout (16x32 logical tile):
// lane l: row = l&15, hi = l>>4; VGPR0-3 hold K = hi*8+{0..7},
// VGPR4-7 hold K = 16+hi*8+{0..7}.  Two ds_load_b128 per lane.
// ---------------------------------------------------------------------------
__device__ __forceinline__ v16h load_frag(const _Float16* base, int ld, int lane) {
  const int m = lane & 15, hi = lane >> 4;
  const _Float16* p = base + m * ld + hi * 8;
  v8h lo = *(const v8h*)(p);
  v8h hv = *(const v8h*)(p + 16);
  v16h r;
#pragma unroll
  for (int i = 0; i < 8; ++i) { r[i] = lo[i]; r[i + 8] = hv[i]; }
  return r;
}

__device__ __forceinline__ v8f wmma32(v16h a, v16h b, v8f c) {
  return __builtin_amdgcn_wmma_f32_16x16x32_f16(false, a, false, b, (short)0, c,
                                                false, false);
}

// Async DMA: global -> LDS, 16 bytes per lane, tracked by ASYNCcnt.
// Generic shared pointer low 32 bits == LDS byte offset (aperture rule).
__device__ __forceinline__ void async_b128(_Float16* lds_dst,
                                           const _Float16* gsrc) {
  unsigned loff = (unsigned)(uintptr_t)lds_dst;
  asm volatile("global_load_async_to_lds_b128 %0, %1, off"
               :: "v"(loff), "v"(gsrc)
               : "memory");
}
__device__ __forceinline__ void wait_async_le8() {
  asm volatile("s_wait_asynccnt 0x8" ::: "memory");
}
__device__ __forceinline__ void wait_async_0() {
  asm volatile("s_wait_asynccnt 0x0" ::: "memory");
}
__device__ __forceinline__ void wait_ds0() {
  asm volatile("s_wait_dscnt 0x0" ::: "memory");
}

// ---------------------------------------------------------------------------
__global__ void cvt_f32_f16(const float* __restrict__ s, _Float16* __restrict__ d,
                            int n) {
  int i = blockIdx.x * blockDim.x + threadIdx.x;
  if (i < n) d[i] = (_Float16)s[i];
}

// ---------------------------------------------------------------------------
// q/k/v projection (64x64 tiles, f16 in / f32 acc, async double-buffered
// staging) + rotary + relayout.  z = 0:q (rotary), 1:k (scale+rotary),
// 2:v (copy, stored TRANSPOSED as (B,H,D,T) so attention can DMA it rowwise).
// ---------------------------------------------------------------------------
__global__ __launch_bounds__(128) void proj_rotary_kernel(
    const _Float16* __restrict__ xh, const _Float16* __restrict__ wqh,
    const _Float16* __restrict__ wkh, const _Float16* __restrict__ wvh,
    const float* __restrict__ sinp, const float* __restrict__ cosp,
    _Float16* __restrict__ qh, _Float16* __restrict__ kh,
    _Float16* __restrict__ vht) {
  __shared__ __align__(16) _Float16 Asb[2 * 64 * LDA];
  __shared__ __align__(16) _Float16 Bsb[2 * 64 * LDA];
  __shared__ __align__(16) float    Cs[64 * LDF];

  const int tid = threadIdx.x, lane = tid & 31, wid = tid >> 5;
  const int m0 = wid * 16, hi = lane >> 4, nn = lane & 15;
  const int nBase = blockIdx.x * 64, mBase = blockIdx.y * 64;
  const int z = blockIdx.z;
  const _Float16* wh = (z == 0) ? wqh : (z == 1) ? wkh : wvh;

  v8f acc[4];
#pragma unroll
  for (int i = 0; i < 4; ++i)
#pragma unroll
    for (int j = 0; j < 8; ++j) acc[i][j] = 0.f;

  auto stage = [&](int kk, int buf) {
    _Float16* Ad = Asb + buf * (64 * LDA);
    _Float16* Bd = Bsb + buf * (64 * LDA);
    for (int e = tid; e < 512; e += 128) {      // 4 iters -> 8 async/wave
      int r = e >> 3, c = (e & 7) << 3;
      async_b128(Ad + r * LDA + c, xh + (size_t)(mBase + r) * EE + kk + c);
      async_b128(Bd + r * LDA + c, wh + (size_t)(nBase + r) * EE + kk + c);
    }
  };

  stage(0, 0);
  for (int kc = 0; kc < EE / 64; ++kc) {
    const int cur = kc & 1;
    if (kc + 1 < EE / 64) { stage((kc + 1) * 64, cur ^ 1); wait_async_le8(); }
    else                  { wait_async_0(); }
    __syncthreads();
    const _Float16* Ac = Asb + cur * (64 * LDA);
    const _Float16* Bc = Bsb + cur * (64 * LDA);
    // batch all fragment loads, then run the WMMA chain back-to-back
    v16h a_lo = load_frag(Ac + m0 * LDA, LDA, lane);
    v16h a_hi = load_frag(Ac + m0 * LDA + 32, LDA, lane);
    v16h bl[4], bh[4];
#pragma unroll
    for (int nt = 0; nt < 4; ++nt) {
      bl[nt] = load_frag(Bc + nt * 16 * LDA, LDA, lane);
      bh[nt] = load_frag(Bc + nt * 16 * LDA + 32, LDA, lane);
    }
#pragma unroll
    for (int nt = 0; nt < 4; ++nt) {
      acc[nt] = wmma32(a_lo, bl[nt], acc[nt]);
      acc[nt] = wmma32(a_hi, bh[nt], acc[nt]);
    }
    __syncthreads();
  }

#pragma unroll
  for (int nt = 0; nt < 4; ++nt)
#pragma unroll
    for (int r = 0; r < 8; ++r)
      Cs[(m0 + r + 8 * hi) * LDF + nt * 16 + nn] = acc[nt][r];
  __syncthreads();

  const float kscale = 0.125f;  // D^-0.5
  if (z == 2) {
    // v: write transposed (B,H,D,T); i indexes t (coalesced), j indexes d
    for (int e = tid; e < 4096; e += 128) {
      int i = e & 63, j = e >> 6;
      int row = mBase + i;
      int b = row / TT, t = row % TT;
      int n = nBase + j;
      int h = n >> 6, d = n & 63;
      vht[(((size_t)(b * HH + h)) * DD + d) * TT + t] = (_Float16)Cs[i * LDF + j];
    }
  } else {
    for (int e = tid; e < 4096; e += 128) {
      int i = e >> 6, j = e & 63;
      int row = mBase + i;
      int b = row / TT, t = row % TT;
      int n = nBase + j;
      int h = n >> 6, d = n & 63;
      float val = Cs[i * LDF + j];
      float partner = Cs[i * LDF + (j ^ 1)];
      float rot = (d & 1) ? partner : -partner;  // even d: -x[d+1], odd: x[d-1]
      float rr = val * cosp[t * DD + d] + rot * sinp[t * DD + d];
      if (z == 1) rr *= kscale;
      ((z == 0) ? qh : kh)[(((size_t)(b * HH + h)) * TT + t) * DD + d] =
          (_Float16)rr;
    }
  }
}

// ---------------------------------------------------------------------------
// Flash-style retention attention per (b,h): 64-row q tiles, async
// double-buffered 64-wide K/V tiles, multiplicative-mask online softmax,
// fused RMS norm.  V comes in pre-transposed (B,H,D,T).
// ---------------------------------------------------------------------------
__global__ __launch_bounds__(128) void attn_kernel(
    const _Float16* __restrict__ qh, const _Float16* __restrict__ kh,
    const _Float16* __restrict__ vht, const float* __restrict__ mask,
    _Float16* __restrict__ aoh) {
  __shared__ __align__(16) _Float16 QPs[64 * LDA];        // Q staging, then P
  __shared__ __align__(16) _Float16 Ksb[2 * 64 * LDA];
  __shared__ __align__(16) _Float16 Vtb[2 * 64 * LDA];    // [d][t'] rows
  __shared__ __align__(16) float    Ss[64 * LDF];
  __shared__ float Mst[64], Lst[64], Scl[64];

  const int tid = threadIdx.x, lane = tid & 31, wid = tid >> 5;
  const int m0 = wid * 16, hi = lane >> 4, nn = lane & 15;
  const int bh = blockIdx.y, b = bh / HH, h = bh % HH;
  const int qt = blockIdx.x;
  const size_t headBase = (size_t)bh * TT * DD;   // same for (T,D) and (D,T)

  auto stage_tile = [&](int kt, int buf) {
    _Float16* Kd = Ksb + buf * (64 * LDA);
    _Float16* Vd = Vtb + buf * (64 * LDA);
    const _Float16* ks = kh + headBase + (size_t)kt * 64 * DD;
    const _Float16* vs = vht + headBase + (size_t)kt * 64;
    for (int e = tid; e < 512; e += 128) {      // 4 iters -> 8 async/wave
      int r = e >> 3, c = (e & 7) << 3;
      async_b128(Kd + r * LDA + c, ks + (size_t)r * DD + c);
      async_b128(Vd + r * LDA + c, vs + (size_t)r * TT + c);
    }
  };

  if (tid < 64) { Mst[tid] = -1e30f; Lst[tid] = 0.f; Scl[tid] = 1.f; }
  for (int e = tid; e < 512; e += 128) {  // Q staging (sync)
    int r = e >> 3, c = (e & 7) << 3;
    *(v8h*)(QPs + r * LDA + c) =
        *(const v8h*)(qh + headBase + (size_t)(qt * 64 + r) * DD + c);
  }
  stage_tile(0, 0);
  __syncthreads();
  v16h aq_lo = load_frag(QPs + m0 * LDA, LDA, lane);
  v16h aq_hi = load_frag(QPs + m0 * LDA + 32, LDA, lane);
  _Float16* Ps = QPs;  // reuse buffer: Q fragments are now in registers

  v8f o[4];
#pragma unroll
  for (int i = 0; i < 4; ++i)
#pragma unroll
    for (int j = 0; j < 8; ++j) o[i][j] = 0.f;

  for (int kt = 0; kt < TT / 64; ++kt) {
    const int cur = kt & 1;
    if (kt + 1 < TT / 64) { stage_tile(kt + 1, cur ^ 1); wait_async_le8(); }
    else                  { wait_async_0(); }
    __syncthreads();
    const _Float16* Kc = Ksb + cur * (64 * LDA);
    const _Float16* Vc = Vtb + cur * (64 * LDA);

    // S = Q K^T  (per-wave 16x64 strip): batch loads, then 8 chained WMMA
    {
      v16h bl[4], bh2[4];
#pragma unroll
      for (int nt = 0; nt < 4; ++nt) {
        bl[nt]  = load_frag(Kc + nt * 16 * LDA, LDA, lane);
        bh2[nt] = load_frag(Kc + nt * 16 * LDA + 32, LDA, lane);
      }
      v8f s[4];
#pragma unroll
      for (int nt = 0; nt < 4; ++nt) {
#pragma unroll
        for (int j = 0; j < 8; ++j) s[nt][j] = 0.f;
        s[nt] = wmma32(aq_lo, bl[nt], s[nt]);
        s[nt] = wmma32(aq_hi, bh2[nt], s[nt]);
      }
#pragma unroll
      for (int nt = 0; nt < 4; ++nt)
#pragma unroll
        for (int r = 0; r < 8; ++r)
          Ss[(m0 + r + 8 * hi) * LDF + nt * 16 + nn] = s[nt][r];
    }
    wait_ds0();

    // online softmax: lane pair (nn, nn+16) covers row m0+nn, 32 cols each
    {
      const int rowT = m0 + nn;
      const int tq = qt * 64 + rowT;
      const int c0 = hi * 32;
      const float* mkp = mask + ((size_t)h * TT + tq) * TT + (size_t)kt * 64 + c0;
      __builtin_prefetch(mkp + 64, 0, 1);   // next k-tile's mask row
      const float* srow = Ss + rowT * LDF + c0;
      float pv[32];
      float mx = -1e30f;
#pragma unroll
      for (int j = 0; j < 32; ++j) {
        float v = srow[j] * mkp[j];
        pv[j] = v;
        mx = fmaxf(mx, v);
      }
      mx = fmaxf(mx, __shfl_xor(mx, 16, 32));
      float oldm = Mst[rowT];
      float newm = fmaxf(oldm, mx);
      float sc = __expf(oldm - newm);
      float sum = 0.f;
      _Float16* prow = Ps + rowT * LDA + c0;
#pragma unroll
      for (int j = 0; j < 32; ++j) {
        float ev = __expf(pv[j] - newm);
        sum += ev;
        prow[j] = (_Float16)ev;
      }
      sum += __shfl_xor(sum, 16, 32);
      if (hi == 0) {
        Mst[rowT] = newm;
        Lst[rowT] = Lst[rowT] * sc + sum;
        Scl[rowT] = sc;
      }
    }
    wait_ds0();

    // rescale O accumulators by this tile's correction factor
#pragma unroll
    for (int r = 0; r < 8; ++r) {
      float sc = Scl[m0 + hi * 8 + r];
#pragma unroll
      for (int nt = 0; nt < 4; ++nt) o[nt][r] *= sc;
    }

    // O += P V  (B-operand columns are Vt rows): batch loads, then WMMA chain
    {
      v16h ap_lo = load_frag(Ps + m0 * LDA, LDA, lane);
      v16h ap_hi = load_frag(Ps + m0 * LDA + 32, LDA, lane);
      v16h bl[4], bh2[4];
#pragma unroll
      for (int nt = 0; nt < 4; ++nt) {
        bl[nt]  = load_frag(Vc + nt * 16 * LDA, LDA, lane);
        bh2[nt] = load_frag(Vc + nt * 16 * LDA + 32, LDA, lane);
      }
#pragma unroll
      for (int nt = 0; nt < 4; ++nt) {
        o[nt] = wmma32(ap_lo, bl[nt], o[nt]);
        o[nt] = wmma32(ap_hi, bh2[nt], o[nt]);
      }
    }
    __syncthreads();
  }

  // finalize: divide by softmax denom, stage to LDS
#pragma unroll
  for (int r = 0; r < 8; ++r) {
    float inv = 1.f / fmaxf(Lst[m0 + hi * 8 + r], 1e-30f);
#pragma unroll
    for (int nt = 0; nt < 4; ++nt)
      Ss[(m0 + r + 8 * hi) * LDF + nt * 16 + nn] = o[nt][r] * inv;
  }
  wait_ds0();

  // fused RMS norm over D, write f16 (B,T,H*D) for the output GEMM
  {
    const int rowT = m0 + nn;
    const int tq = qt * 64 + rowT;
    const int c0 = hi * 32;
    const float* srow = Ss + rowT * LDF + c0;
    float ssq = 0.f;
#pragma unroll
    for (int j = 0; j < 32; ++j) { float v = srow[j]; ssq += v * v; }
    ssq += __shfl_xor(ssq, 16, 32);
    float rs = rsqrtf(ssq * (1.f / (float)DD) + EPSV);
    _Float16* op = aoh + ((size_t)b * TT + tq) * EE + (size_t)h * DD + c0;
#pragma unroll
    for (int j = 0; j < 32; ++j) op[j] = (_Float16)(srow[j] * rs);
  }
}

// ---------------------------------------------------------------------------
// Final GEMM: out = a @ wo.T  (f16 in, f32 out), async double-buffered tiles.
// ---------------------------------------------------------------------------
__global__ __launch_bounds__(128) void out_gemm_kernel(
    const _Float16* __restrict__ ah, const _Float16* __restrict__ woh,
    float* __restrict__ out) {
  __shared__ __align__(16) _Float16 Asb[2 * 64 * LDA];
  __shared__ __align__(16) _Float16 Bsb[2 * 64 * LDA];
  const int tid = threadIdx.x, lane = tid & 31, wid = tid >> 5;
  const int m0 = wid * 16, hi = lane >> 4, nn = lane & 15;
  const int nBase = blockIdx.x * 64, mBase = blockIdx.y * 64;

  v8f acc[4];
#pragma unroll
  for (int i = 0; i < 4; ++i)
#pragma unroll
    for (int j = 0; j < 8; ++j) acc[i][j] = 0.f;

  auto stage = [&](int kk, int buf) {
    _Float16* Ad = Asb + buf * (64 * LDA);
    _Float16* Bd = Bsb + buf * (64 * LDA);
    for (int e = tid; e < 512; e += 128) {
      int r = e >> 3, c = (e & 7) << 3;
      async_b128(Ad + r * LDA + c, ah + (size_t)(mBase + r) * EE + kk + c);
      async_b128(Bd + r * LDA + c, woh + (size_t)(nBase + r) * EE + kk + c);
    }
  };

  stage(0, 0);
  for (int kc = 0; kc < EE / 64; ++kc) {
    const int cur = kc & 1;
    if (kc + 1 < EE / 64) { stage((kc + 1) * 64, cur ^ 1); wait_async_le8(); }
    else                  { wait_async_0(); }
    __syncthreads();
    const _Float16* Ac = Asb + cur * (64 * LDA);
    const _Float16* Bc = Bsb + cur * (64 * LDA);
    v16h a_lo = load_frag(Ac + m0 * LDA, LDA, lane);
    v16h a_hi = load_frag(Ac + m0 * LDA + 32, LDA, lane);
    v16h bl[4], bh[4];
#pragma unroll
    for (int nt = 0; nt < 4; ++nt) {
      bl[nt] = load_frag(Bc + nt * 16 * LDA, LDA, lane);
      bh[nt] = load_frag(Bc + nt * 16 * LDA + 32, LDA, lane);
    }
#pragma unroll
    for (int nt = 0; nt < 4; ++nt) {
      acc[nt] = wmma32(a_lo, bl[nt], acc[nt]);
      acc[nt] = wmma32(a_hi, bh[nt], acc[nt]);
    }
    __syncthreads();
  }
#pragma unroll
  for (int nt = 0; nt < 4; ++nt)
#pragma unroll
    for (int r = 0; r < 8; ++r)
      out[(size_t)(mBase + m0 + r + 8 * hi) * EE + nBase + nt * 16 + nn] =
          acc[nt][r];
}

// ---------------------------------------------------------------------------
extern "C" void kernel_launch(void* const* d_in, const int* in_sizes, int n_in,
                              void* d_out, int out_size, void* d_ws,
                              size_t ws_size, hipStream_t stream) {
  (void)in_sizes; (void)n_in; (void)out_size; (void)ws_size;
  const float* x    = (const float*)d_in[0];
  const float* sinp = (const float*)d_in[1];
  const float* cosp = (const float*)d_in[2];
  const float* mask = (const float*)d_in[3];
  const float* wq   = (const float*)d_in[4];
  const float* wk   = (const float*)d_in[5];
  const float* wv   = (const float*)d_in[6];
  const float* wo   = (const float*)d_in[7];
  float* out = (float*)d_out;

  const size_t NX = (size_t)BB * TT * EE;  // 4M elems
  const size_t NW = (size_t)EE * EE;       // 1M elems
  _Float16* p   = (_Float16*)d_ws;
  _Float16* xh  = p; p += NX;
  _Float16* wqh = p; p += NW;
  _Float16* wkh = p; p += NW;
  _Float16* wvh = p; p += NW;
  _Float16* woh = p; p += NW;
  _Float16* qh  = p; p += NX;
  _Float16* kh  = p; p += NX;
  _Float16* vht = p; p += NX;   // (B,H,D,T)
  _Float16* aoh = p; p += NX;

  cvt_f32_f16<<<(int)(NX / 256), 256, 0, stream>>>(x, xh, (int)NX);
  cvt_f32_f16<<<(int)(NW / 256), 256, 0, stream>>>(wq, wqh, (int)NW);
  cvt_f32_f16<<<(int)(NW / 256), 256, 0, stream>>>(wk, wkh, (int)NW);
  cvt_f32_f16<<<(int)(NW / 256), 256, 0, stream>>>(wv, wvh, (int)NW);
  cvt_f32_f16<<<(int)(NW / 256), 256, 0, stream>>>(wo, woh, (int)NW);

  dim3 gp(EE / 64, (BB * TT) / 64, 3);
  proj_rotary_kernel<<<gp, 128, 0, stream>>>(xh, wqh, wkh, wvh, sinp, cosp, qh,
                                             kh, vht);
  dim3 ga(TT / 64, BB * HH);
  attn_kernel<<<ga, 128, 0, stream>>>(qh, kh, vht, mask, aoh);

  dim3 go(EE / 64, (BB * TT) / 64);
  out_gemm_kernel<<<go, 128, 0, stream>>>(aoh, woh, out);
}
